// CausalSelfAttention_48232482734167
// MI455X (gfx1250) — compile-verified, matrix operand order checked by reading, not run on hardware
//
#include <hip/hip_runtime.h>
#include <hip/hip_bf16.h>
#include <math.h>

// ---------------------------------------------------------------------------
// CDNA5 (gfx1250) causal self-attention, bf16 WMMA with fp32 accumulation.
// B=2, T=2048, C=1024, 16 heads, head_dim=64.
// ---------------------------------------------------------------------------

typedef __attribute__((ext_vector_type(16))) __bf16 v16bf;
typedef __attribute__((ext_vector_type(8)))  float  v8f;

#define B_  2
#define T_  2048
#define C_  1024
#define NH_ 16
#define HD_ 64

// fp32 -> bf16 via the target's native convert (gfx1250 has bf16 hardware;
// __bf16 is a legal type here — the WMMA builtins take v16bf).
__device__ __forceinline__ unsigned short f2bf(float f) {
    return __builtin_bit_cast(unsigned short, static_cast<__bf16>(f));
}
__device__ __forceinline__ unsigned pack2bf(float lo, float hi) {
    return (unsigned)f2bf(lo) | ((unsigned)f2bf(hi) << 16);
}

// K-offset of element pair (e=0) for VGPR v of a 16-bit 16x32 A-frag
// (ISA 7.12.2): lanes 0-15 (half=0): V0..3 -> K 0..7, V4..7 -> K 16..23;
// lanes 16-31 (half=1): V0..3 -> K 8..15, V4..7 -> K 24..31.
__device__ __forceinline__ int kfrag(int v, int half) {
    return ((v >> 2) << 4) | (half << 3) | ((v & 3) << 1);
}

union Frag16 { v16bf bf; unsigned u[8]; };

// Load a 16x32 bf16 fragment when the K dimension is contiguous in memory.
// Works for A-frags (lane = row) and B-frags (lane = column, source stored
// K-contiguous per column). Each VGPR is one dword (2 packed bf16).
__device__ __forceinline__ v16bf load_frag_bf16(const unsigned short* base, int half) {
    Frag16 r;
#pragma unroll
    for (int v = 0; v < 8; ++v)
        r.u[v] = *reinterpret_cast<const unsigned*>(base + kfrag(v, half));
    return r.bf;
}

// Same, but converting from fp32 source on the fly (pairs -> packed cvt).
__device__ __forceinline__ v16bf load_frag_f32(const float* base, int half) {
    Frag16 r;
#pragma unroll
    for (int v = 0; v < 8; ++v) {
        int k = kfrag(v, half);
        r.u[v] = pack2bf(base[k], base[k + 1]);
    }
    return r.bf;
}

// ---------------------------------------------------------------------------
// Generic WMMA GEMM: C[M][N] (fp32) = A[M][K] @ B[K][N] (B fp32 row-major).
// Block = 128 threads = 4 waves. Block tile 128x64; wave tile 32x64
// (2 A-frags x 4 B-frags = 8 wmma per 32-K step -> 2x the WMMA/byte of a
// 16x64 tile). B k-slab (32x64) staged in LDS as bf16, column-major with a
// 34-ushort pitch (17-dword stride = conflict-free banks) so B-frag reads
// are single ds dwords.
// ---------------------------------------------------------------------------
template <bool ABF16>
__global__ void __launch_bounds__(128)
gemm_wmma(const void* __restrict__ Ap, const float* __restrict__ Bm,
          float* __restrict__ Cm, int M, int N, int K) {
    __shared__ unsigned short ldsB[64 * 34];   // [col][k(+pad)]
    const int tid  = threadIdx.x;
    const int lane = tid & 31, wave = tid >> 5;
    const int half = lane >> 4, l16 = lane & 15;
    const int ncol0 = blockIdx.x * 64;
    const int mrow0 = blockIdx.y * 128 + wave * 32;

    v8f acc[2][4];
#pragma unroll
    for (int mf = 0; mf < 2; ++mf)
#pragma unroll
        for (int t = 0; t < 4; ++t) acc[mf][t] = (v8f){};

    const float*          Af32 = (const float*)Ap;
    const unsigned short* Abf  = (const unsigned short*)Ap;
    const size_t arow0 = (size_t)(mrow0 + l16) * K;
    const size_t arow1 = (size_t)(mrow0 + 16 + l16) * K;

    for (int k0 = 0; k0 < K; k0 += 32) {
        // cooperative stage of B[k0:k0+32][ncol0:ncol0+64] -> bf16 LDS.
        // 1024 k-pairs; each thread converts a pair and stores one dword.
#pragma unroll
        for (int i = 0; i < 8; ++i) {
            int p  = i * 128 + tid;
            int c  = p & 63;                  // coalesced along columns
            int kp = p >> 6;                  // 0..15 (k pair)
            float f0 = Bm[(size_t)(k0 + 2 * kp)     * N + ncol0 + c];
            float f1 = Bm[(size_t)(k0 + 2 * kp + 1) * N + ncol0 + c];
            *reinterpret_cast<unsigned*>(&ldsB[c * 34 + 2 * kp]) = pack2bf(f0, f1);
        }
        // speculative prefetch of next k-slab (gfx1250 global_prefetch)
        if (tid < 64 && k0 + 32 < K)
            __builtin_prefetch(Bm + (size_t)(k0 + 32 + (tid & 31)) * N +
                                   ncol0 + (tid >> 5) * 32, 0, 0);
        __syncthreads();

        v16bf a0, a1;
        if constexpr (ABF16) {
            a0 = load_frag_bf16(Abf + arow0 + k0, half);
            a1 = load_frag_bf16(Abf + arow1 + k0, half);
        } else {
            a0 = load_frag_f32(Af32 + arow0 + k0, half);
            a1 = load_frag_f32(Af32 + arow1 + k0, half);
        }

#pragma unroll
        for (int t = 0; t < 4; ++t) {
            v16bf b = load_frag_bf16(ldsB + (t * 16 + l16) * 34, half);
            acc[0][t] = __builtin_amdgcn_wmma_f32_16x16x32_bf16(
                false, a0, false, b, (short)0, acc[0][t], false, false);
            acc[1][t] = __builtin_amdgcn_wmma_f32_16x16x32_bf16(
                false, a1, false, b, (short)0, acc[1][t], false, false);
        }
        __syncthreads();
    }

#pragma unroll
    for (int mf = 0; mf < 2; ++mf)
#pragma unroll
        for (int t = 0; t < 4; ++t)
#pragma unroll
            for (int r = 0; r < 8; ++r) {
                int rg = mrow0 + mf * 16 + half * 8 + r;  // C/D: row = half*8+r
                int cg = ncol0 + t * 16 + l16;            // col = lane&15
                Cm[(size_t)rg * N + cg] = acc[mf][t][r];
            }
}

// ---------------------------------------------------------------------------
// RoPE (reference's exact pair-rotate + concat-freq variant) applied to q,k,
// convert q/k/v to bf16. One thread per (even d) pair: float2 loads, packed
// dword stores. q,k -> [B*H][T][64]; v -> transposed [B*H][64][T] so flash
// attention's P*V B-fragments are K-contiguous dword loads.
// ---------------------------------------------------------------------------
__global__ void __launch_bounds__(256)
rope_pack(const float* __restrict__ qkv, unsigned short* __restrict__ qb,
          unsigned short* __restrict__ kb, unsigned short* __restrict__ vt) {
    int idx = blockIdx.x * 256 + threadIdx.x;      // over B*H*T*32 pairs
    int d0 = (idx & 31) * 2;
    int t  = (idx >> 5) & (T_ - 1);
    int bh = idx >> 16;                            // T*32 = 2^16 per (b,h)
    int b  = bh >> 4, h = bh & 15;

    size_t row = (size_t)(b * T_ + t) * (3 * C_);
    int    col = h * HD_ + d0;
    float2 q2 = *(const float2*)(qkv + row + col);
    float2 k2 = *(const float2*)(qkv + row + C_ + col);
    float2 v2 = *(const float2*)(qkv + row + 2 * C_ + col);

    // emb index d -> inv_freq[d mod 32]; inv_freq[j] = 10000^(-2j/64)
    const float lnk = -0.28782313662425572f;       // -ln(10000)/32
    float sn0, cs0, sn1, cs1;
    sincosf((float)t * __expf(lnk * (float)(d0 & 31)),       &sn0, &cs0);
    sincosf((float)t * __expf(lnk * (float)((d0 + 1) & 31)), &sn1, &cs1);
    // x_rot[2i] = -x[2i+1] ; x_rot[2i+1] = x[2i]
    float qo0 = q2.x * cs0 - q2.y * sn0;
    float qo1 = q2.y * cs1 + q2.x * sn1;
    float ko0 = k2.x * cs0 - k2.y * sn0;
    float ko1 = k2.y * cs1 + k2.x * sn1;

    size_t o = (size_t)bh * T_ * HD_ + (size_t)t * HD_ + d0;
    *reinterpret_cast<unsigned*>(qb + o) = pack2bf(qo0, qo1);
    *reinterpret_cast<unsigned*>(kb + o) = pack2bf(ko0, ko1);
    size_t vo = (size_t)bh * HD_ * T_ + (size_t)d0 * T_ + t;
    vt[vo]      = f2bf(v2.x);
    vt[vo + T_] = f2bf(v2.y);
}

// ---------------------------------------------------------------------------
// Flash attention (causal, online softmax).  One wave per 16-row query tile;
// block = 4 waves = 64 query rows of one (b,h).  32 keys per step:
//   4 wmma (S = Q K^T), mask+scale, LDS bounce C->A layout, per-row softmax
//   (row stats live in a lane pair -> one shfl_xor(16)), 4 wmma (O += P V).
// Output y written bf16 in [B*T][C] layout for the projection GEMM.
// ---------------------------------------------------------------------------
__global__ void __launch_bounds__(128)
flash_attn(const unsigned short* __restrict__ qg,
           const unsigned short* __restrict__ kg,
           const unsigned short* __restrict__ vtg,
           unsigned short* __restrict__ yb) {
    __shared__ float ldsS[4][16][32];   // per-wave S scratch, 8 KB
    const int lane = threadIdx.x & 31, wave = threadIdx.x >> 5;
    const int half = lane >> 4, l16 = lane & 15;
    const int bh = blockIdx.x / (T_ / 64);
    const int qt = blockIdx.x % (T_ / 64);
    const int b  = bh >> 4, h = bh & 15;
    const int q_row0 = qt * 64 + wave * 16;

    const unsigned short* qbh = qg  + (size_t)bh * T_ * HD_;
    const unsigned short* kbh = kg  + (size_t)bh * T_ * HD_;
    const unsigned short* vbh = vtg + (size_t)bh * HD_ * T_;

    const v16bf qa0 = load_frag_bf16(qbh + (size_t)(q_row0 + l16) * HD_,      half);
    const v16bf qa1 = load_frag_bf16(qbh + (size_t)(q_row0 + l16) * HD_ + 32, half);

    v8f o[4];
#pragma unroll
    for (int t = 0; t < 4; ++t) o[t] = (v8f){};
    float mrow = -1e30f, lrow = 0.0f;       // state for row l16 (lane pair)
    const float scale = 0.125f;             // 1/sqrt(64)
    const int q_hi = q_row0 + 15;

    for (int kv0 = 0; kv0 <= q_hi; kv0 += 32) {
        // ---- S = Q K^T over hd=64, two 16-key column tiles ----
        v8f s0 = (v8f){}, s1 = (v8f){};
        {
            v16bf kb = load_frag_bf16(kbh + (size_t)(kv0 + l16) * HD_, half);
            s0 = __builtin_amdgcn_wmma_f32_16x16x32_bf16(false, qa0, false, kb, 0, s0, false, false);
            kb = load_frag_bf16(kbh + (size_t)(kv0 + l16) * HD_ + 32, half);
            s0 = __builtin_amdgcn_wmma_f32_16x16x32_bf16(false, qa1, false, kb, 0, s0, false, false);
            kb = load_frag_bf16(kbh + (size_t)(kv0 + 16 + l16) * HD_, half);
            s1 = __builtin_amdgcn_wmma_f32_16x16x32_bf16(false, qa0, false, kb, 0, s1, false, false);
            kb = load_frag_bf16(kbh + (size_t)(kv0 + 16 + l16) * HD_ + 32, half);
            s1 = __builtin_amdgcn_wmma_f32_16x16x32_bf16(false, qa1, false, kb, 0, s1, false, false);
        }
        // ---- causal mask + scale, spill C-layout tile to LDS ----
#pragma unroll
        for (int r = 0; r < 8; ++r) {
            int rg = q_row0 + half * 8 + r;
            float v0 = (kv0 + l16      <= rg) ? s0[r] * scale : -1e30f;
            float v1 = (kv0 + 16 + l16 <= rg) ? s1[r] * scale : -1e30f;
            ldsS[wave][half * 8 + r][l16]      = v0;
            ldsS[wave][half * 8 + r][16 + l16] = v1;
        }
        // (same-wave DS ops are in-order; no barrier needed)

        // ---- online softmax for row l16; read back in A-frag order ----
        float sv[16];
        float rmax = -1e30f;
#pragma unroll
        for (int v = 0; v < 8; ++v) {
            int k = kfrag(v, half);
            sv[2 * v]     = ldsS[wave][l16][k];
            sv[2 * v + 1] = ldsS[wave][l16][k + 1];
            rmax = fmaxf(rmax, fmaxf(sv[2 * v], sv[2 * v + 1]));
        }
        rmax = fmaxf(rmax, __shfl_xor(rmax, 16, 32));
        float mnew  = fmaxf(mrow, rmax);
        float alpha = __expf(mrow - mnew);
        float psum  = 0.0f;
        Frag16 pa;
#pragma unroll
        for (int v = 0; v < 8; ++v) {
            float p0 = __expf(sv[2 * v]     - mnew);
            float p1 = __expf(sv[2 * v + 1] - mnew);
            psum += p0 + p1;
            pa.u[v] = pack2bf(p0, p1);
        }
        psum += __shfl_xor(psum, 16, 32);
        lrow = lrow * alpha + psum;
        mrow = mnew;

        // ---- rescale O (broadcast alpha from lane holding each row) ----
#pragma unroll
        for (int r = 0; r < 8; ++r) {
            float ar = __shfl(alpha, half * 8 + r, 32);
#pragma unroll
            for (int t = 0; t < 4; ++t) o[t][r] *= ar;
        }
        // ---- O += P V  (V transposed: K-contiguous per output column) ----
#pragma unroll
        for (int t = 0; t < 4; ++t) {
            v16bf vb = load_frag_bf16(vbh + (size_t)(t * 16 + l16) * T_ + kv0, half);
            o[t] = __builtin_amdgcn_wmma_f32_16x16x32_bf16(
                false, pa.bf, false, vb, 0, o[t], false, false);
        }
    }

    // ---- normalize and store y (bf16, [B*T][C]) ----
    float linv = 1.0f / lrow;
#pragma unroll
    for (int r = 0; r < 8; ++r) {
        float lr = __shfl(linv, half * 8 + r, 32);
        int rg = q_row0 + half * 8 + r;
        size_t yrow = ((size_t)b * T_ + rg) * C_;
#pragma unroll
        for (int t = 0; t < 4; ++t)
            yb[yrow + h * HD_ + t * 16 + l16] = f2bf(o[t][r] * lr);
    }
}

// ---------------------------------------------------------------------------
extern "C" void kernel_launch(void* const* d_in, const int* in_sizes, int n_in,
                              void* d_out, int out_size, void* d_ws, size_t ws_size,
                              hipStream_t stream) {
    const float* x      = (const float*)d_in[0];   // [B*T][C] fp32
    const float* w_qkv  = (const float*)d_in[1];   // [C][3C] fp32
    const float* w_proj = (const float*)d_in[2];   // [C][C]  fp32
    float*       out    = (float*)d_out;           // [B*T][C] fp32

    char* ws = (char*)d_ws;
    // Workspace layout (peak ~75.5 MB):
    //   [0, 50.3MB)        qkv fp32 (later overlaid by y bf16, 8.4MB)
    //   [50.3, 58.7MB)     q bf16  [B*H][T][64]
    //   [58.7, 67.1MB)     k bf16  [B*H][T][64]
    //   [67.1, 75.5MB)     v bf16 transposed [B*H][64][T]
    float*          qkv = (float*)ws;
    unsigned short* qb  = (unsigned short*)(ws + (size_t)B_ * T_ * 3 * C_ * 4);
    unsigned short* kb  = qb + (size_t)B_ * NH_ * T_ * HD_;
    unsigned short* vt  = kb + (size_t)B_ * NH_ * T_ * HD_;
    unsigned short* yb  = (unsigned short*)ws;     // overlays qkv (safe: stream order)

    const int M = B_ * T_;                         // 4096

    // 1) qkv = x @ w_qkv           (fp32 A converted in-register)
    gemm_wmma<false><<<dim3(3 * C_ / 64, M / 128), 128, 0, stream>>>(
        x, w_qkv, qkv, M, 3 * C_, C_);

    // 2) RoPE + bf16 pack (q,k row-major; v transposed)
    rope_pack<<<(B_ * NH_ * T_ * HD_ / 2) / 256, 256, 0, stream>>>(qkv, qb, kb, vt);

    // 3) causal flash attention -> y (bf16)
    flash_attn<<<B_ * NH_ * (T_ / 64), 128, 0, stream>>>(qb, kb, vt, yb);

    // 4) out = y @ w_proj          (bf16 A direct dword fragment loads)
    gemm_wmma<true><<<dim3(C_ / 64, M / 128), 128, 0, stream>>>(
        yb, w_proj, out, M, C_, C_);
}
